// TreeEncoder_20040317403463
// MI455X (gfx1250) — matrix operand order repaired and checked by reading, not compiled, not run on hardware
//
#include <hip/hip_runtime.h>
#include <hip/hip_bf16.h>

// ---------------------------------------------------------------------------
// TreeLSTM (2 layers, depth 15, H=1024) for gfx1250 (MI455X).
// Heavy math runs on v_wmma_f32_16x16x32_bf16 (bf16 A/B, fp32 accumulate).
// LDS staging uses global_load_async_to_lds_b128 (ASYNCcnt-tracked DMA path).
// ---------------------------------------------------------------------------

typedef __bf16 bf16;
typedef __attribute__((ext_vector_type(16))) __bf16 v16bf;
typedef __attribute__((ext_vector_type(8)))  float  v8f;
typedef __attribute__((ext_vector_type(4)))  int    i4;

#define TREE_DEPTH 15
#define N_NODES    32767
#define HID        1024

#if defined(__AMDGCN__) && __has_builtin(__builtin_amdgcn_global_load_async_to_lds_b128)
#define USE_ASYNC_LDS 1
#else
#define USE_ASYNC_LDS 0
#endif

static __device__ __forceinline__ float sigmoidf_(float x) {
  return 1.0f / (1.0f + __expf(-x));
}

// Preorder position of heap node i in a complete binary tree of depth 15.
// Walk root->node: left step adds 1, right step adds 2^(D-l-1).
static __device__ __forceinline__ int preorder_pos(int i) {
  unsigned node = (unsigned)(i + 1);
  int L = 31 - __clz(node);   // level of node (0 = root)
  int pos = 0;
  for (int l = 0; l < L; ++l) {
    int bit = (node >> (L - 1 - l)) & 1;
    pos += bit ? (1 << (TREE_DEPTH - 1 - l)) : 1;
  }
  return pos;
}

// ---------------------------------------------------------------------------
// Pack per-layer weights into Wp[5120, K] (bf16), K = Din + 2*H.
//   rows [0,3072):  iou  -> [ W_iou | Uiou[0] | Uiou[1] ]
//   rows [3072,4096): f_l -> [ W_f  | Uf[0]   |   0     ]
//   rows [4096,5120): f_r -> [ W_f  |   0     | Uf[1]   ]
// bias bp[5120] = [b_iou, b_f, b_f]
// ---------------------------------------------------------------------------
__global__ __launch_bounds__(256)
void packw_kernel(const float* __restrict__ W, const float* __restrict__ b,
                  const float* __restrict__ Uiou, const float* __restrict__ Uf,
                  int Din, bf16* __restrict__ Wp, float* __restrict__ bp) {
  const int H = HID;
  const int K = Din + 2 * H;
  unsigned idx = blockIdx.x * 256u + threadIdx.x;
  unsigned total = (unsigned)(5 * H) * (unsigned)K;
  if (idx >= total) return;
  int r = idx / K, c = idx % K;
  float v = 0.0f;
  if (r < 3 * H) {
    if (c < Din)            v = W[(size_t)r * Din + c];
    else if (c < Din + H)   v = Uiou[(size_t)r * H + (c - Din)];
    else                    v = Uiou[(size_t)3 * H * H + (size_t)r * H + (c - Din - H)];
  } else if (r < 4 * H) {
    int rf = r - 3 * H;
    if (c < Din)            v = W[(size_t)(3 * H + rf) * Din + c];
    else if (c < Din + H)   v = Uf[(size_t)rf * H + (c - Din)];
  } else {
    int rf = r - 4 * H;
    if (c < Din)            v = W[(size_t)(3 * H + rf) * Din + c];
    else if (c >= Din + H)  v = Uf[(size_t)H * H + (size_t)rf * H + (c - Din - H)];
  }
  Wp[idx] = (bf16)v;
  if (c == 0) {
    float bv = (r < 3 * H) ? b[r] : ((r < 4 * H) ? b[3 * H + (r - 3 * H)]
                                                 : b[3 * H + (r - 4 * H)]);
    bp[r] = bv;
  }
}

// Embedding gather -> bf16.  grid = N_NODES blocks x 256 threads (EMB=256).
__global__ __launch_bounds__(256)
void embed_kernel(const int* __restrict__ tokens, const float* __restrict__ emb,
                  bf16* __restrict__ xbf) {
  int i = blockIdx.x, j = threadIdx.x;
  xbf[(size_t)i * 256 + j] = (bf16)emb[(size_t)tokens[i] * 256 + j];
}

// Pack per-level GEMM input rows  V[r] = [ x_i | h_{2i+1} | h_{2i+2} ].
__global__ __launch_bounds__(256)
void packv_kernel(const bf16* __restrict__ xsrc, int Din,
                  const bf16* __restrict__ hbf, bf16* __restrict__ V,
                  int start, int n) {
  const int K = Din + 2 * HID;
  unsigned idx = blockIdx.x * 256u + threadIdx.x;
  unsigned total = (unsigned)n * (unsigned)K;
  if (idx >= total) return;
  int r = idx / K, c = idx % K;
  int i = start + r;
  bf16 val;
  if (c < Din)             val = xsrc[(size_t)i * Din + c];
  else if (c < Din + HID)  val = hbf[(size_t)(2 * i + 1) * HID + (c - Din)];
  else                     val = hbf[(size_t)(2 * i + 2) * HID + (c - Din - HID)];
  V[idx] = val;
}

// ---------------------------------------------------------------------------
// GEMM:  C[M, Nout] = A[M, K] @ B[Nout, K]^T + bias,  A/B bf16, C fp32.
// Block tile 128x128, 8 waves (4x2), each wave computes 32x64 via 2x4 WMMA
// tiles (8 v_wmma per K-step).  K % 32 == 0, Nout % 128 == 0.
// Out-of-range A rows are clamped to row M-1 (their C rows are never stored).
// ---------------------------------------------------------------------------
#define BM 128
#define BN 128
#define BK 32

union FragU { v16bf v; uint4 q[2]; };

static __device__ __forceinline__ void stage16(const bf16* gsrc, bf16* ldst) {
#if USE_ASYNC_LDS
  __builtin_amdgcn_global_load_async_to_lds_b128(
      (__attribute__((address_space(1))) i4*)(void*)gsrc,
      (__attribute__((address_space(3))) i4*)(void*)ldst, 0, 0);
#else
  *(uint4*)ldst = *(const uint4*)gsrc;
#endif
}

static __device__ __forceinline__ void stage_fence() {
#if USE_ASYNC_LDS
#if __has_builtin(__builtin_amdgcn_s_wait_asynccnt)
  __builtin_amdgcn_s_wait_asynccnt(0);
#else
  asm volatile("s_wait_asynccnt 0x0" ::: "memory");
#endif
#endif
}

__global__ __launch_bounds__(256)
void gemm_bf16_kernel(const bf16* __restrict__ A, int lda, int M, int K,
                      const bf16* __restrict__ B, int ldb,
                      const float* __restrict__ bias,
                      float* __restrict__ C, int ldc) {
  __shared__ bf16 sA[BM * BK];   // 8 KB
  __shared__ bf16 sB[BN * BK];   // 8 KB
  const int tid  = threadIdx.x;
  const int wave = tid >> 5;
  const int lane = tid & 31;
  const int hi   = lane >> 4;       // half-wave select
  const int ln   = lane & 15;
  const int bm = blockIdx.y * BM;
  const int bn = blockIdx.x * BN;
  const int wm = (wave >> 1) * 32;  // wave M offset within block (0..96)
  const int wn = (wave & 1) * 64;   // wave N offset within block (0/64)

  // Staging coordinates: each thread moves 2x b128 for A and 2x for B.
  const int sr = tid >> 1;              // 0..127
  const int sc = (tid & 1) * 16;        // 0 or 16
  int agr = bm + sr; if (agr >= M) agr = M - 1;   // clamp ragged edge
  const bf16* aSrc = A + (size_t)agr * lda + sc;
  const bf16* bSrc = B + (size_t)(bn + sr) * ldb + sc;

  v8f acc[2][4] = {};

  for (int k0 = 0; k0 < K; k0 += BK) {
    stage16(aSrc + k0,     &sA[sr * BK + sc]);
    stage16(aSrc + k0 + 8, &sA[sr * BK + sc + 8]);
    stage16(bSrc + k0,     &sB[sr * BK + sc]);
    stage16(bSrc + k0 + 8, &sB[sr * BK + sc + 8]);
    stage_fence();
    __syncthreads();

    // A fragment (16x32 bf16): lane holds M=ln; elems 0..7 -> K=hi*8..+7,
    // elems 8..15 -> K=16+hi*8..+7  (ISA 16-bit A layout) -> two b128 loads.
    FragU aF[2], bF[4];
#pragma unroll
    for (int tm = 0; tm < 2; ++tm) {
      int mrow = wm + tm * 16 + ln;
      aF[tm].q[0] = *(const uint4*)&sA[mrow * BK + hi * 8];
      aF[tm].q[1] = *(const uint4*)&sA[mrow * BK + 16 + hi * 8];
    }
    // B fragment (32x16 bf16): lane holds N=ln; elems 0..7 -> K=hi*8..+7, elems 8..15 -> K=16+hi*8..+7.
#pragma unroll
    for (int tn = 0; tn < 4; ++tn) {
      int nrow = wn + tn * 16 + ln;
      bF[tn].q[0] = *(const uint4*)&sB[nrow * BK + hi * 8];
      bF[tn].q[1] = *(const uint4*)&sB[nrow * BK + 16 + hi * 8];
    }

#pragma unroll
    for (int tm = 0; tm < 2; ++tm)
#pragma unroll
      for (int tn = 0; tn < 4; ++tn)
        acc[tm][tn] = __builtin_amdgcn_wmma_f32_16x16x32_bf16(
            false, aF[tm].v, false, bF[tn].v, (short)0, acc[tm][tn],
            false, false);
    __syncthreads();
  }

  // Epilogue: C/D layout -> VGPR v: M = v + hi*8, N = ln.
#pragma unroll
  for (int tm = 0; tm < 2; ++tm)
#pragma unroll
    for (int tn = 0; tn < 4; ++tn) {
      int col = bn + wn + tn * 16 + ln;
      float bb = bias[col];
#pragma unroll
      for (int v = 0; v < 8; ++v) {
        int row = bm + wm + tm * 16 + hi * 8 + v;
        if (row < M) C[(size_t)row * ldc + col] = acc[tm][tn][v] + bb;
      }
    }
}

// ---------------------------------------------------------------------------
// Gate nonlinearities + cell/hidden update for one level (fp32 math).
// G rows: [i(1024) | o(1024) | u(1024) | f_l(1024) | f_r(1024)] (iou only for leaves)
// ---------------------------------------------------------------------------
__global__ __launch_bounds__(256)
void combine_kernel(const float* __restrict__ G, int Gw, int start, int n,
                    int isLeaf, const float* __restrict__ cChild,
                    float* __restrict__ cCur, bf16* __restrict__ hbf,
                    float* __restrict__ out, int layer, int isRoot,
                    int writeAnn) {
  unsigned idx = blockIdx.x * 256u + threadIdx.x;
  if (idx >= (unsigned)n * HID) return;
  int r = idx >> 10, j = idx & 1023;
  const float* g = G + (size_t)r * Gw;
  float ig = sigmoidf_(g[j]);
  float og = sigmoidf_(g[HID + j]);
  float ug = tanhf(g[2 * HID + j]);
  float c;
  if (isLeaf) {
    c = ig * ug;
  } else {
    float fl = sigmoidf_(g[3 * HID + j]);
    float fr = sigmoidf_(g[4 * HID + j]);
    c = ig * ug + fl * cChild[(size_t)(2 * r) * HID + j]
                + fr * cChild[(size_t)(2 * r + 1) * HID + j];
  }
  float h = og * tanhf(c);
  cCur[(size_t)r * HID + j] = c;
  int i = start + r;
  hbf[(size_t)i * HID + j] = (bf16)h;
  if (writeAnn)  // layer 1: h goes straight to its preorder slot in d_out
    out[(size_t)preorder_pos(i) * HID + j] = h;
  if (isRoot && r == 0) {
    out[(size_t)(N_NODES + layer) * HID + j] = h;       // hiddens[layer]
    out[(size_t)(N_NODES + 2 + layer) * HID + j] = c;   // cell_states[layer]
  }
}

// ---------------------------------------------------------------------------
extern "C" void kernel_launch(void* const* d_in, const int* in_sizes, int n_in,
                              void* d_out, int out_size, void* d_ws,
                              size_t ws_size, hipStream_t stream) {
  (void)in_sizes; (void)n_in; (void)out_size; (void)ws_size;
  const int*   tokens = (const int*)d_in[0];
  const float* emb    = (const float*)d_in[1];
  const float* W0     = (const float*)d_in[2];
  const float* b0     = (const float*)d_in[3];
  const float* Uiou0  = (const float*)d_in[4];
  const float* Uf0    = (const float*)d_in[5];
  const float* W1     = (const float*)d_in[6];
  const float* b1     = (const float*)d_in[7];
  const float* Uiou1  = (const float*)d_in[8];
  const float* Uf1    = (const float*)d_in[9];
  float* out = (float*)d_out;

  // Workspace carve-up (~590 MB total).
  char* p = (char*)d_ws;
  auto carve = [&](size_t bytes) {
    void* r = (void*)p;
    p += (bytes + 255) & ~(size_t)255;
    return r;
  };
  bf16*  Wp0  = (bf16*) carve((size_t)5120 * 2304 * 2);
  bf16*  Wp1  = (bf16*) carve((size_t)5120 * 3072 * 2);
  float* bp0  = (float*)carve((size_t)5120 * 4);
  float* bp1  = (float*)carve((size_t)5120 * 4);
  bf16*  xbf  = (bf16*) carve((size_t)N_NODES * 256 * 2);
  bf16*  h0bf = (bf16*) carve((size_t)N_NODES * HID * 2);
  bf16*  h1bf = (bf16*) carve((size_t)N_NODES * HID * 2);
  float* cA   = (float*)carve((size_t)16384 * HID * 4);
  float* cB   = (float*)carve((size_t)16384 * HID * 4);
  bf16*  V    = (bf16*) carve((size_t)8192 * 3072 * 2);
  float* G    = (float*)carve((size_t)16384 * 3072 * 4);  // >= 8192*5120*4

  // Pack weights (bf16) + biases.
  {
    unsigned t0 = 5120u * 2304u;
    packw_kernel<<<(t0 + 255) / 256, 256, 0, stream>>>(W0, b0, Uiou0, Uf0, 256,
                                                       Wp0, bp0);
    unsigned t1 = 5120u * 3072u;
    packw_kernel<<<(t1 + 255) / 256, 256, 0, stream>>>(W1, b1, Uiou1, Uf1, 1024,
                                                       Wp1, bp1);
  }
  embed_kernel<<<N_NODES, 256, 0, stream>>>(tokens, emb, xbf);

  for (int layer = 0; layer < 2; ++layer) {
    const int   Din   = layer ? 1024 : 256;
    const int   Kfull = Din + 2 * HID;
    const bf16* xsrc  = layer ? h0bf : xbf;
    bf16*       hbf   = layer ? h1bf : h0bf;
    const bf16* Wp    = layer ? Wp1 : Wp0;
    const float* bp   = layer ? bp1 : bp0;
    const int writeAnn = (layer == 1);

    for (int lvl = TREE_DEPTH - 1; lvl >= 0; --lvl) {
      const int n = 1 << lvl, start = n - 1;
      float* cCur   = (lvl & 1) ? cB : cA;
      float* cChild = (lvl & 1) ? cA : cB;
      if (lvl == TREE_DEPTH - 1) {
        // Leaves: no children -> only iou rows (3072), only x columns (Din).
        dim3 grid(3072 / BN, (n + BM - 1) / BM);
        gemm_bf16_kernel<<<grid, 256, 0, stream>>>(
            xsrc + (size_t)start * Din, Din, n, Din, Wp, Kfull, bp, G, 3072);
        combine_kernel<<<((unsigned)n * HID + 255) / 256, 256, 0, stream>>>(
            G, 3072, start, n, /*leaf*/1, nullptr, cCur, hbf, out, layer,
            /*root*/0, writeAnn);
      } else {
        unsigned tv = (unsigned)n * (unsigned)Kfull;
        packv_kernel<<<(tv + 255) / 256, 256, 0, stream>>>(xsrc, Din, hbf, V,
                                                           start, n);
        dim3 grid(5120 / BN, (n + BM - 1) / BM);
        gemm_bf16_kernel<<<grid, 256, 0, stream>>>(V, Kfull, n, Kfull, Wp,
                                                   Kfull, bp, G, 5120);
        combine_kernel<<<((unsigned)n * HID + 255) / 256, 256, 0, stream>>>(
            G, 5120, start, n, /*leaf*/0, cChild, cCur, hbf, out, layer,
            /*root*/(lvl == 0), writeAnn);
      }
    }
  }
}
